// GeneAttentionLayer_87093346828515
// MI455X (gfx1250) — matrix operand order repaired and checked by reading, not compiled, not run on hardware
//
#include <hip/hip_runtime.h>

typedef __attribute__((ext_vector_type(16))) _Float16 v16h;
typedef __attribute__((ext_vector_type(8)))  _Float16 v8h;
typedef __attribute__((ext_vector_type(8)))  float    v8f;

#define WMMA16(a,b,c) __builtin_amdgcn_wmma_f32_16x16x32_f16(false,(a),false,(b),(short)0,(c),false,false)

constexpr int NB   = 4;
constexpr int NSEQ = 2048;
constexpr int NH   = 8;
constexpr int DH   = 64;
constexpr int DM   = NH * DH;   // 512
constexpr int KT   = 64;        // key tile
constexpr float SM_SCALE = 0.125f;  // 1/sqrt(64)
constexpr float L1_EPS   = 1e-8f;

// A-operand fragment (16-bit A 16x32): halves 0..7 <- p[0..7], halves 8..15 <- p[16..23]
__device__ __forceinline__ v16h frag_gap(const _Float16* p) {
  v8h lo = *(const v8h*)p;
  v8h hi = *(const v8h*)(p + 16);
  v16h r;
#pragma unroll
  for (int i = 0; i < 8; ++i) { r[i] = lo[i]; r[i + 8] = hi[i]; }
  return r;
}

// B-operand fragment: 16 contiguous halves
__device__ __forceinline__ v16h frag_seq(const _Float16* p) {
  v8h lo = *(const v8h*)p;
  v8h hi = *(const v8h*)(p + 8);
  v16h r;
#pragma unroll
  for (int i = 0; i < 8; ++i) { r[i] = lo[i]; r[i + 8] = hi[i]; }
  return r;
}

// gfx1250 async DMA: global -> LDS, 16 bytes per enabled lane, tracked by ASYNCcnt.
// LDS byte address = low 32 bits of the generic pointer (ISA aperture truncation).
__device__ __forceinline__ void async_ld_b128(const void* g, void* l) {
  unsigned lds = (unsigned)(unsigned long long)(uintptr_t)l;
  asm volatile("global_load_async_to_lds_b128 %0, %1, off"
               :: "v"(lds), "v"(g)
               : "memory");
}
__device__ __forceinline__ void wait_async8() {
  asm volatile("s_wait_asynccnt 0x8" ::: "memory");
}
__device__ __forceinline__ void wait_async0() {
  asm volatile("s_wait_asynccnt 0x0" ::: "memory");
}

// ---------------------------------------------------------------------------
// Stage 1: fused attention (softmax * mask, L1 renorm) -> X[B*N, 512] (f16)
// grid = (N/128, B*H), block = 256 (8 waves; each wave owns 16 query rows)
// ---------------------------------------------------------------------------
__global__ __launch_bounds__(256) void attn_fused(
    const float* __restrict__ Q, const float* __restrict__ K,
    const float* __restrict__ V, const float* __restrict__ Msk,
    _Float16* __restrict__ X) {
  __shared__ __align__(16) _Float16 sK[KT * DH];       // [key][d]   (A of S^T)
  __shared__ __align__(16) _Float16 sVt[DH * KT];      // [d][key]   (A of O^T)
  __shared__ __align__(16) _Float16 sP[8][16 * KT];    // per wave: P [q][key]
  __shared__ __align__(16) float    sM[2][128 * KT];   // mask tiles, double-buffered

  const int tid   = threadIdx.x;
  const int lane  = tid & 31;
  const int wave  = tid >> 5;
  const int lhalf = lane >> 4;     // 0 | 1
  const int l16   = lane & 15;

  const int bh  = blockIdx.y;
  const int b   = bh >> 3, h = bh & 7;
  const int qb0 = blockIdx.x * 128;
  const int q0  = qb0 + wave * 16;
  const int qg  = q0 + l16;

  // Q as the B operand of S^T = K x Q^T  (K-dim = d, column = q)
  const float* Qrow = Q + (((size_t)b * NSEQ + qg) * NH + h) * DH;
  v16h qB[2];
#pragma unroll
  for (int c = 0; c < 2; ++c) {
    const float* p = Qrow + c * 32 + lhalf * 16;
#pragma unroll
    for (int i = 0; i < 16; ++i) qB[c][i] = (_Float16)p[i];
  }

  v8f acc[4] = {};                  // O^T tiles: rows = d (4x16), cols = q
  float mrow = -1e30f, lrow = 0.0f; // per-lane (column q) running stats

  // Prologue: async-stage mask tile 0 (128 q-rows x 64 keys, f32) into buffer 0.
  // 8 float4 transfers per thread per tile -> 8 outstanding async ops per wave.
  {
#pragma unroll
    for (int i = 0; i < 8; ++i) {
      int e4   = i * 256 + tid;          // float4 index within tile
      int qloc = e4 >> 4;                // 0..127
      int k4   = e4 & 15;                // float4 within 64-key row
      const float* g = Msk + ((size_t)b * NSEQ + (qb0 + qloc)) * NSEQ + k4 * 4;
      async_ld_b128(g, &sM[0][e4 * 4]);
    }
  }

  for (int kb = 0; kb < NSEQ; kb += KT) {
    const int cur = (kb / KT) & 1;
    __syncthreads();   // prior iteration done reading sK/sVt/sP

    // cooperative load+convert of K/V tiles (float4 reads), prefetch next tile
#pragma unroll
    for (int i = 0; i < 4; ++i) {
      int e4  = i * 256 + tid;           // float4 index, 0..1023
      int key = e4 >> 4;                 // 0..63
      int d4  = e4 & 15;                 // float4 within head dim
      size_t g = (((size_t)b * NSEQ + (kb + key)) * NH + h) * DH + d4 * 4;
      float4 kv = *(const float4*)(K + g);
      float4 vv = *(const float4*)(V + g);
      if (kb + KT < NSEQ && i == 0) {
        __builtin_prefetch(K + g + (size_t)KT * NH * DH, 0, 1);
        __builtin_prefetch(V + g + (size_t)KT * NH * DH, 0, 1);
      }
      int d = d4 * 4;
      sK[key * DH + d + 0] = (_Float16)kv.x;
      sK[key * DH + d + 1] = (_Float16)kv.y;
      sK[key * DH + d + 2] = (_Float16)kv.z;
      sK[key * DH + d + 3] = (_Float16)kv.w;
      sVt[(d + 0) * KT + key] = (_Float16)vv.x;
      sVt[(d + 1) * KT + key] = (_Float16)vv.y;
      sVt[(d + 2) * KT + key] = (_Float16)vv.z;
      sVt[(d + 3) * KT + key] = (_Float16)vv.w;
    }

    // async-stage next mask tile into the other buffer, then drain current tile
    if (kb + KT < NSEQ) {
#pragma unroll
      for (int i = 0; i < 8; ++i) {
        int e4   = i * 256 + tid;
        int qloc = e4 >> 4;
        int k4   = e4 & 15;
        const float* g = Msk + ((size_t)b * NSEQ + (qb0 + qloc)) * NSEQ
                         + (kb + KT) + k4 * 4;
        async_ld_b128(g, &sM[cur ^ 1][e4 * 4]);
      }
      wait_async8();   // async loads complete in order: current tile is done
    } else {
      wait_async0();
    }
    __syncthreads();   // all waves' shares of sK/sVt/sM[cur] are visible

    // S^T = K x Q^T : 4 tiles of 16(keys) x 16(q), f32 accum
    v8f S[4] = {};
#pragma unroll
    for (int t = 0; t < 4; ++t) {
#pragma unroll
      for (int c = 0; c < 2; ++c) {
        v16h aK = frag_seq(&sK[(t * 16 + l16) * DH + c * 32 + lhalf * 16]);
        S[t] = WMMA16(aK, qB[c], S[t]);
      }
    }

    // online softmax * mask; stats live per-lane (column q = l16)
    float mnew = mrow;
#pragma unroll
    for (int t = 0; t < 4; ++t)
#pragma unroll
      for (int r = 0; r < 8; ++r) {
        float s = S[t][r] * SM_SCALE;
        S[t][r] = s;
        mnew = fmaxf(mnew, s);
      }
    mnew = fmaxf(mnew, __shfl_xor(mnew, 16));
    float corr = __expf(mrow - mnew);
    float lsum = 0.0f;
    const float* mtile = &sM[cur][(wave * 16 + l16) * KT];
#pragma unroll
    for (int t = 0; t < 4; ++t)
#pragma unroll
      for (int r = 0; r < 8; ++r) {
        int key = t * 16 + r + lhalf * 8;    // C-layout row index = key
        float p = __expf(S[t][r] - mnew) * mtile[key];
        lsum += p;
        sP[wave][l16 * KT + key] = (_Float16)p;   // P stored [q][key]
      }
    lsum += __shfl_xor(lsum, 16);
    mrow = mnew;
    lrow = lrow * corr + lsum;
#pragma unroll
    for (int t = 0; t < 4; ++t)
#pragma unroll
      for (int r = 0; r < 8; ++r) acc[t][r] *= corr;

    __syncthreads();   // order sP scatter-writes vs cross-lane fragment reads

    // O^T += V^T x P : M = d (4 tiles), K-dim = key (2 chunks), N = q
#pragma unroll
    for (int dt = 0; dt < 4; ++dt) {
#pragma unroll
      for (int c = 0; c < 2; ++c) {
        v16h aV = frag_seq(&sVt[(dt * 16 + l16) * KT + c * 32 + lhalf * 16]);
        v16h bP = frag_seq(&sP[wave][l16 * KT + c * 32 + lhalf * 16]);
        acc[dt] = WMMA16(aV, bP, acc[dt]);
      }
    }
  }

  // L1 normalize and emit X[b*N+q, h*64+d] as f16 for the projection GEMM
  float inv = 1.0f / (lrow + L1_EPS);
  _Float16* Xrow = X + ((size_t)b * NSEQ + qg) * DM + h * DH;
#pragma unroll
  for (int dt = 0; dt < 4; ++dt)
#pragma unroll
    for (int r = 0; r < 8; ++r) {
      int d = dt * 16 + r + lhalf * 8;
      Xrow[d] = (_Float16)(acc[dt][r] * inv);
    }
}

// ---------------------------------------------------------------------------
// Stage 2: out[8192,512] = X_f16 @ W^T + b  (WMMA GEMM, W slice in LDS)
// grid = (8192/128, 512/64), block = 256
// ---------------------------------------------------------------------------
__global__ __launch_bounds__(256) void out_proj(
    const _Float16* __restrict__ X, const float* __restrict__ W,
    const float* __restrict__ bias, float* __restrict__ out) {
  __shared__ __align__(16) _Float16 sW[64 * DM];   // W[n0+n][k], 64 KB

  const int tid   = threadIdx.x;
  const int lane  = tid & 31;
  const int wave  = tid >> 5;
  const int lhalf = lane >> 4;
  const int l16   = lane & 15;
  const int n0    = blockIdx.y * 64;
  const int m0    = blockIdx.x * 128 + wave * 16;

#pragma unroll 8
  for (int i = 0; i < 32; ++i) {
    int e4 = i * 256 + tid;                        // float4 index, 0..8191
    const float4 w4 = *(const float4*)
        (W + (size_t)(n0 + (e4 >> 7)) * DM + (e4 & 127) * 4);
    int e = e4 * 4;
    sW[e + 0] = (_Float16)w4.x;
    sW[e + 1] = (_Float16)w4.y;
    sW[e + 2] = (_Float16)w4.z;
    sW[e + 3] = (_Float16)w4.w;
  }
  __syncthreads();

  const _Float16* Xrow = X + (size_t)(m0 + l16) * DM;
  v8f acc[4] = {};
  for (int kc = 0; kc < DM; kc += 32) {
    v16h aX = frag_seq(Xrow + kc + lhalf * 16);    // A: rows = out rows, K = k
#pragma unroll
    for (int t = 0; t < 4; ++t) {                  // B: (k, n) = W[n][k]
      v16h bW = frag_seq(&sW[(t * 16 + l16) * DM + kc + lhalf * 16]);
      acc[t] = WMMA16(aX, bW, acc[t]);
    }
  }

#pragma unroll
  for (int t = 0; t < 4; ++t) {
    int col = n0 + t * 16 + l16;
    float bv = bias[col];
#pragma unroll
    for (int r = 0; r < 8; ++r) {
      int row = m0 + r + lhalf * 8;
      out[(size_t)row * DM + col] = acc[t][r] + bv;
    }
  }
}

// ---------------------------------------------------------------------------
extern "C" void kernel_launch(void* const* d_in, const int* in_sizes, int n_in,
                              void* d_out, int out_size, void* d_ws, size_t ws_size,
                              hipStream_t stream) {
  (void)in_sizes; (void)n_in; (void)out_size; (void)ws_size;
  const float* Q    = (const float*)d_in[0];
  const float* K    = (const float*)d_in[1];
  const float* V    = (const float*)d_in[2];
  const float* Msk  = (const float*)d_in[3];
  const float* Wout = (const float*)d_in[4];
  const float* bout = (const float*)d_in[5];
  _Float16*    X    = (_Float16*)d_ws;            // 8192*512 f16 = 8 MB scratch
  float*       out  = (float*)d_out;

  dim3 g1(NSEQ / 128, NB * NH);                   // (16, 32)
  attn_fused<<<g1, 256, 0, stream>>>(Q, K, V, Msk, X);

  dim3 g2((NB * NSEQ) / 128, DM / 64);            // (64, 8)
  out_proj<<<g2, 256, 0, stream>>>(X, Wout, bout, out);
}
